// MultiDiscreteACTLayer_29240137351782
// MI455X (gfx1250) — matrix operand order repaired and checked by reading, not compiled, not run on hardware
//
#include <hip/hip_runtime.h>
#include <math.h>

typedef _Float16 v16h __attribute__((ext_vector_type(16)));
typedef float    v8f  __attribute__((ext_vector_type(8)));
typedef float    v4f  __attribute__((ext_vector_type(4)));

#define D_DIM 256
#define N_SC 16
#define SC_CAP 2
#define MAX_USERS 8
#define POW_DIM 10
#define NCOLS 208           // 8*16 sc + 8*10 pow
#define NTILES 13           // 208 / 16
#define WH_PITCH 272        // half-pitch per col (544B, 32B aligned v16h loads)
#define L_PITCH 209         // float-pitch per row for logits (odd -> conflict-free scalar reads)
#define ROWS_PER_WAVE 16
#define ROWS_PER_BLOCK 128
#define W_BYTES (NCOLS * WH_PITCH * 2)               // 113152
#define LG_BYTES (ROWS_PER_BLOCK * L_PITCH * 4)      // 107008
#define SMEM_BYTES (W_BYTES + LG_BYTES)              // 220160 < 320KB WGP LDS

__device__ __forceinline__ float rng_u01(unsigned a) {
  a ^= a >> 16; a *= 0x7feb352dU;
  a ^= a >> 15; a *= 0x846ca68bU;
  a ^= a >> 16;
  return (float)(a >> 8) * (1.0f / 16777216.0f) + 1.0e-7f;
}

__global__ void act_fused_kernel(const float* __restrict__ x,
                                 const int* __restrict__ avail,
                                 const float* __restrict__ W_sc,
                                 const float* __restrict__ b_sc,
                                 const float* __restrict__ W_pow,
                                 const float* __restrict__ b_pow,
                                 float* __restrict__ out,
                                 int B) {
  extern __shared__ char smem_raw[];
  _Float16* Wh = (_Float16*)smem_raw;            // [col][k] f16, pitch WH_PITCH
  float* lg = (float*)(smem_raw + W_BYTES);      // [lrow][col] f32 logits, pitch L_PITCH

  const int tid  = threadIdx.x;
  const int lane = tid & 31;
  const int wave = tid >> 5;

  // ---- Phase 0: stage packed weights (256 x 208) into LDS as f16 [col][k] ----
  for (int idx = tid; idx < NCOLS * D_DIM; idx += blockDim.x) {
    const int col = idx >> 8;       // /256
    const int d   = idx & 255;
    float v;
    if (col < MAX_USERS * N_SC) {
      const int u = col >> 4, k = col & 15;
      v = W_sc[(u * D_DIM + d) * N_SC + k];
    } else {
      const int cc = col - MAX_USERS * N_SC;
      const int u = cc / POW_DIM, j = cc - u * POW_DIM;
      v = W_pow[(u * D_DIM + d) * POW_DIM + j];
    }
    Wh[col * WH_PITCH + d] = (_Float16)v;
  }
  __syncthreads();

  // ---- Phase 1: f16 WMMA GEMM (f32 accum), 16 rows/wave x 208 cols ----
  const int rowBase = blockIdx.x * ROWS_PER_BLOCK + wave * ROWS_PER_WAVE;
  const int mRow = lane & 15;     // M (A rows) / N (B cols)
  const int half = lane >> 4;     // K-half selector per ISA 16-bit layouts

  v8f acc[NTILES];
  const v8f vzero = {};
#pragma unroll
  for (int n = 0; n < NTILES; ++n) acc[n] = vzero;

  const float* xr = x + (size_t)(rowBase + mRow) * D_DIM;

  for (int kk = 0; kk < D_DIM / 32; ++kk) {     // 8 K-chunks of 32
    const int base_lo = 32 * kk + 16 * half;    // lanes 0-15: K 0-7 ; lanes 16-31: K 16-23
    const int base_hi = base_lo + 8;            // lanes 0-15: K 8-15; lanes 16-31: K 24-31
    v4f q0 = *(const v4f*)(xr + base_lo);
    v4f q1 = *(const v4f*)(xr + base_lo + 4);
    v4f q2 = *(const v4f*)(xr + base_hi);
    v4f q3 = *(const v4f*)(xr + base_hi + 4);
    __builtin_prefetch(xr + base_lo + 64, 0, 1);   // global_prefetch_b8, 2 chunks ahead

    v16h a;
#pragma unroll
    for (int j = 0; j < 4; ++j) {
      a[j]      = (_Float16)q0[j];
      a[4 + j]  = (_Float16)q1[j];
      a[8 + j]  = (_Float16)q2[j];
      a[12 + j] = (_Float16)q3[j];
    }

#pragma unroll
    for (int n = 0; n < NTILES; ++n) {
      // B 32x16: lane=N, 16 consecutive K halves; lanes 16-31 take K 16..31 of chunk
      const _Float16* bp = Wh + (n * 16 + mRow) * WH_PITCH + 32 * kk + 16 * half;
      v16h b = *(const v16h*)bp;
      acc[n] = __builtin_amdgcn_wmma_f32_16x16x32_f16(
          /*neg_a=*/false, a, /*neg_b=*/false, b,
          /*c_mod=*/(short)0, acc[n], /*reuse_a=*/false, /*reuse_b=*/false);
    }
  }

  // ---- Phase 2: spill logits (+bias) into LDS: logits[lrow 0..127][col 0..207] ----
#pragma unroll
  for (int n = 0; n < NTILES; ++n) {
    const int col = n * 16 + mRow;
    const float bias = (col < MAX_USERS * N_SC) ? b_sc[col] : b_pow[col - MAX_USERS * N_SC];
#pragma unroll
    for (int i = 0; i < 8; ++i) {
      const int lrow = wave * 16 + i + 8 * half;   // C/D layout: VGPR i -> M=i (+8 for hi half)
      lg[lrow * L_PITCH + col] = acc[n][i] + bias;
    }
  }
  __syncthreads();

  // ---- Phase 3: per-row sequential sampling (one lane per row, lanes 0..15) ----
  if (lane < 16) {
    const int lrow = wave * 16 + lane;
    const int row  = rowBase + lane;
    const float* L = lg + lrow * L_PITCH;
    unsigned sc_stat = 0;           // 2 bits per subcarrier channel, cap 2
    float logp_sum = 0.0f;

    // subcarrier heads (sequential capacity mask)
    for (int u = 0; u < MAX_USERS; ++u) {
      const float av = (float)avail[row * 16 + u];
      float m = -3.0e38f;
#pragma unroll
      for (int c = 0; c < N_SC; ++c) {
        float l = L[u * N_SC + c];
        if (((sc_stat >> (2 * c)) & 3u) >= SC_CAP) l = -1.0e10f;
        m = fmaxf(m, l);
      }
      float s = 0.0f;
      int amax = 0;
      float gbest = -3.0e38f;
#pragma unroll
      for (int c = 0; c < N_SC; ++c) {
        float l = L[u * N_SC + c];
        if (((sc_stat >> (2 * c)) & 3u) >= SC_CAP) l = -1.0e10f;
        s += __expf(l - m);
        const float uu = rng_u01(((unsigned)row * 0x9E3779B9u) ^
                                 ((unsigned)(u * 0x85EBCA6B + c * 0xC2B2AE35 + 0x27D4EB2F)));
        const float g = l - __logf(-__logf(uu));   // Gumbel-argmax categorical
        if (g > gbest) { gbest = g; amax = c; }
      }
      float lsel = L[u * N_SC + amax];
      if (((sc_stat >> (2 * amax)) & 3u) >= SC_CAP) lsel = -1.0e10f;
      const float logp = (lsel - m - __logf(s)) * av;
      const int action = (av > 0.0f) ? amax : -1;
      if (action >= 0) sc_stat += 1u << (2 * amax);
      out[(size_t)row * 16 + u] = (float)action;
      logp_sum += logp;
    }

    // power heads (no mask)
    for (int u = 0; u < MAX_USERS; ++u) {
      const float* Lp = L + MAX_USERS * N_SC + u * POW_DIM;
      const float av = (float)avail[row * 16 + 8 + u];
      float m = -3.0e38f;
#pragma unroll
      for (int c = 0; c < POW_DIM; ++c) m = fmaxf(m, Lp[c]);
      float s = 0.0f;
      int amax = 0;
      float gbest = -3.0e38f;
#pragma unroll
      for (int c = 0; c < POW_DIM; ++c) {
        const float l = Lp[c];
        s += __expf(l - m);
        const float uu = rng_u01(((unsigned)row * 0xB5297A4Du) ^
                                 ((unsigned)((u + 8) * 0x68E31DA4 + c * 0x1B56C4E9 + 999)));
        const float g = l - __logf(-__logf(uu));
        if (g > gbest) { gbest = g; amax = c; }
      }
      const float logp = (Lp[amax] - m - __logf(s)) * av;
      const int action = (av > 0.0f) ? amax : -1;
      out[(size_t)row * 16 + 8 + u] = (float)action;
      logp_sum += logp;
    }

    // outputs: [actions (B,16)] [logp_sum (B,1)] [available (B,16)] flat
    out[(size_t)B * 16 + row] = logp_sum;
#pragma unroll
    for (int j = 0; j < 16; ++j)
      out[(size_t)B * 17 + (size_t)row * 16 + j] = (float)avail[row * 16 + j];
  }
}

extern "C" void kernel_launch(void* const* d_in, const int* in_sizes, int n_in,
                              void* d_out, int out_size, void* d_ws, size_t ws_size,
                              hipStream_t stream) {
  const float* x     = (const float*)d_in[0];
  const int*   avail = (const int*)d_in[1];
  const float* W_sc  = (const float*)d_in[2];
  const float* b_sc  = (const float*)d_in[3];
  const float* W_pow = (const float*)d_in[4];
  const float* b_pow = (const float*)d_in[5];
  float* out = (float*)d_out;

  const int B = in_sizes[0] / D_DIM;        // 131072
  const int grid = B / ROWS_PER_BLOCK;      // 1024 blocks of 8 waves
  hipLaunchKernelGGL(act_fused_kernel, dim3(grid), dim3(256), SMEM_BYTES, stream,
                     x, avail, W_sc, b_sc, W_pow, b_pow, out, B);
}
